// Attention_90718299226831
// MI455X (gfx1250) — matrix operand order repaired and checked by reading, compiled, not run on hardware
//
#include <hip/hip_runtime.h>

// ---- problem constants ----
#define BB 4
#define NN 1568          // tokens per batch; 1568 = 98*16 = 49*32
#define CC 768
#define HH 12
#define DD 64
#define NT16 98          // NN/16
#define NT32 49          // NN/32
#define TOK (BB*NN)      // 6272
#define SCALE 0.125f     // DD^-0.5

typedef float v8f  __attribute__((ext_vector_type(8)));
typedef __bf16 v16bf __attribute__((ext_vector_type(16)));

union ABFrag { uint4 u[2]; v16bf v; };

__device__ __forceinline__ v8f wmma_bf16(const ABFrag& a, const ABFrag& b, v8f c) {
  return __builtin_amdgcn_wmma_f32_16x16x32_bf16(false, a.v, false, b.v, (short)0, c,
                                                 false, false);
}

__device__ __forceinline__ unsigned short f2bf(float f) {
  unsigned u = __float_as_uint(f);
  u = (u + 0x7FFFu + ((u >> 16) & 1u)) >> 16;   // round-to-nearest-even
  return (unsigned short)u;
}

__device__ __forceinline__ v8f zero8() {
  v8f z = {0.f,0.f,0.f,0.f,0.f,0.f,0.f,0.f};
  return z;
}

// ---------------------------------------------------------------------------
// fp32 -> bf16 row-major convert (x)
// ---------------------------------------------------------------------------
__global__ __launch_bounds__(256) void cvt_x_kernel(const float* __restrict__ x,
                                                    unsigned short* __restrict__ xb,
                                                    int n) {
  int i = (blockIdx.x * 256 + threadIdx.x) * 4;
  if (i >= n) return;
  float4 f = *(const float4*)&x[i];
  uint2 u;
  u.x = (unsigned)f2bf(f.x) | ((unsigned)f2bf(f.y) << 16);
  u.y = (unsigned)f2bf(f.z) | ((unsigned)f2bf(f.w) << 16);
  *(uint2*)&xb[i] = u;
}

// ---------------------------------------------------------------------------
// Pack a row-major fp32 weight [K x Nc] into WMMA B-fragment bf16 layout:
//   dst[(kstep*(Nc/16) + ntile)*512 + lane*16 + j]
//   lane = (n&15) + bit3(k&31)*16 ; j = (k&7) + bit4(k&31)*8   (ISA 7.12.2, 16-bit B 32x16)
// ---------------------------------------------------------------------------
__global__ __launch_bounds__(256) void pack_w_kernel(const float* __restrict__ src,
                                                     unsigned short* __restrict__ dst,
                                                     int K, int Nc) {
  int i = blockIdx.x * 256 + threadIdx.x;
  if (i >= K * Nc) return;
  int k = i / Nc, n = i - k * Nc;
  int kk = k & 31;
  int lane = (n & 15) + (((kk >> 3) & 1) << 4);
  int j = (kk & 7) + ((kk >> 4) << 3);
  size_t off = ((size_t)((k >> 5) * (Nc >> 4) + (n >> 4)) << 9) + lane * 16 + j;
  dst[off] = f2bf(src[(size_t)k * Nc + n]);
}

// ---------------------------------------------------------------------------
// GEMM: A bf16 row-major [M x K] (staged in LDS), B pre-swizzled fragments.
// Block = 256 threads (8 waves), tile 128(M) x 128(N), BK = 32.
// Wave (wm 0..3, wn 0..1) computes 32x64 via 2x4 WMMA register block.
// EPI==0: QKV epilogue -> scatter into QA / KB / VB fragment buffers.
// EPI==1: proj epilogue -> +bias, fp32 row-major out.
// ---------------------------------------------------------------------------
template<int EPI>
__global__ __launch_bounds__(256)
void gemm_kernel(const unsigned short* __restrict__ A,
                 const unsigned short* __restrict__ Bw,
                 int Ncols, int Kdim,
                 const float* __restrict__ bias0,   // q_bias / proj_b
                 const float* __restrict__ bias1,   // v_bias / unused
                 float* __restrict__ outf,
                 unsigned short* __restrict__ qa,
                 unsigned short* __restrict__ kb,
                 unsigned short* __restrict__ vb) {
  __shared__ __align__(16) unsigned short ldsA[128 * 32];
  const int tid = threadIdx.x, lane = tid & 31, warp = tid >> 5;
  const int wm = warp >> 1, wn = warp & 1;
  const int rb = blockIdx.y * 128, nb = blockIdx.x * 128;
  const int ksteps = Kdim >> 5, ntn = Ncols >> 4;
  const int hi8 = (lane >> 4) << 3, nl = lane & 15, ho = (lane >> 4) << 3;

  v8f acc[2][4];
#pragma unroll
  for (int mt = 0; mt < 2; ++mt)
#pragma unroll
    for (int nt = 0; nt < 4; ++nt) acc[mt][nt] = zero8();

  for (int ks = 0; ks < ksteps; ++ks) {
    // stage A tile (128 rows x 32 bf16) into LDS, 16B per thread-chunk
#pragma unroll
    for (int i = 0; i < 2; ++i) {
      int idx = tid + i * 256;
      int row = idx >> 2, c = idx & 3;
      *(uint4*)&ldsA[row * 32 + c * 8] =
          *(const uint4*)&A[(size_t)(rb + row) * Kdim + ks * 32 + c * 8];
    }
    __syncthreads();

    ABFrag af[2];
    {
      int r0 = wm * 32 + nl;
      const uint4* p0 = (const uint4*)&ldsA[r0 * 32 + ho];
      af[0].u[0] = p0[0]; af[0].u[1] = p0[2];
      const uint4* p1 = (const uint4*)&ldsA[(r0 + 16) * 32 + ho];
      af[1].u[0] = p1[0]; af[1].u[1] = p1[2];
    }
    const int ct = (nb >> 4) + wn * 4;
#pragma unroll
    for (int nt = 0; nt < 4; ++nt) {
      ABFrag bf;
      const uint4* bp = (const uint4*)&Bw[(((size_t)ks * ntn + ct + nt) << 9) + lane * 16];
      bf.u[0] = bp[0]; bf.u[1] = bp[1];
      acc[0][nt] = wmma_bf16(af[0], bf, acc[0][nt]);
      acc[1][nt] = wmma_bf16(af[1], bf, acc[1][nt]);
    }
    __syncthreads();
  }

  // ---- epilogue ----
#pragma unroll
  for (int mt = 0; mt < 2; ++mt) {
    int row0 = rb + wm * 32 + mt * 16;
#pragma unroll
    for (int nt = 0; nt < 4; ++nt) {
      int col = nb + wn * 64 + nt * 16 + nl;
      if (EPI == 1) {
        float pb = bias0[col];
#pragma unroll
        for (int r = 0; r < 8; ++r) {
          int tok = row0 + r + hi8;
          outf[(size_t)tok * CC + col] = acc[mt][nt][r] + pb;
        }
      } else {
        int sec = col >= 2 * CC ? 2 : (col >= CC ? 1 : 0);
        int cc = col - sec * CC;
        int h = cc >> 6, d = cc & 63;
        float bias = (sec == 0) ? bias0[cc] : (sec == 2 ? bias1[cc] : 0.f);
#pragma unroll
        for (int r = 0; r < 8; ++r) {
          int tok = row0 + r + hi8;
          int b = tok / NN;
          int n = tok - b * NN;
          int bh = b * HH + h;
          float v = acc[mt][nt][r] + bias;
          if (sec == 0) {               // Q -> A-fragment layout, pre-scaled
            v *= SCALE;
            int kk = d & 31;
            int la = (n & 15) + ((kk & 8) ? 16 : 0);
            int jj = (kk & 7) + ((kk >> 4) << 3);
            size_t off = ((((size_t)(bh * NT16 + (n >> 4)) * 2 + (d >> 5)) << 5) + la) * 16 + jj;
            qa[off] = f2bf(v);
          } else if (sec == 1) {        // K^T -> B-fragment layout [dstep][ntile]
            int kk = d & 31;
            int la = (n & 15) + (((kk >> 3) & 1) << 4);
            size_t off = ((((size_t)(bh * 2 + (d >> 5)) * NT16 + (n >> 4)) << 5) + la) * 16 + ((kk & 7) + ((kk >> 4) << 3));
            kb[off] = f2bf(v);
          } else {                      // V -> B-fragment layout [kstep][dtile]
            int kk = n & 31;
            int la = (d & 15) + (((kk >> 3) & 1) << 4);
            size_t off = ((((size_t)(bh * NT32 + (n >> 5)) * 4 + (d >> 4)) << 5) + la) * 16 + ((kk & 7) + ((kk >> 4) << 3));
            vb[off] = f2bf(v);
          }
        }
      }
    }
  }
}

// ---------------------------------------------------------------------------
// Flash attention. One wave owns a 16-row Q strip of one (b,h); loops over
// 49 key chunks of 32 keys.  Per chunk: 4 WMMA (S) + online softmax with
// fully-unrolled shfl_xor row-max + WMMA row-sum (P x ones) + 4 WMMA (PV).
// All 8 K/V fragments are loaded at the top of the chunk so V latency is
// hidden behind the softmax VALU section.
// ---------------------------------------------------------------------------
__global__ __launch_bounds__(256)
void attn_kernel(const unsigned short* __restrict__ qa,
                 const unsigned short* __restrict__ kbuf,
                 const unsigned short* __restrict__ vbuf,
                 unsigned short* __restrict__ attnb) {
  __shared__ __align__(16) unsigned short pbuf[8][512];
  const int lane = threadIdx.x & 31, warp = threadIdx.x >> 5;
  const int w = blockIdx.x * 8 + warp;          // 0 .. B*H*98-1
  const int bh = w / NT16, mt = w - bh * NT16;
  const int b = bh / HH, h = bh - b * HH;
  const int hi8 = (lane >> 4) << 3, nl = lane & 15;
  unsigned short* pb = pbuf[warp];

  ABFrag qf0, qf1;
  {
    const uint4* p = (const uint4*)&qa[(((size_t)(bh * NT16 + mt) * 2 + 0) << 9) + lane * 16];
    qf0.u[0] = p[0]; qf0.u[1] = p[1];
    const uint4* p2 = (const uint4*)&qa[(((size_t)(bh * NT16 + mt) * 2 + 1) << 9) + lane * 16];
    qf1.u[0] = p2[0]; qf1.u[1] = p2[1];
  }

  ABFrag ones;   // all-ones B fragment: rowsum(P) = P x ones via the matrix pipe
  ones.u[0] = make_uint4(0x3F803F80u, 0x3F803F80u, 0x3F803F80u, 0x3F803F80u);
  ones.u[1] = ones.u[0];

  v8f o[4];
#pragma unroll
  for (int dt = 0; dt < 4; ++dt) o[dt] = zero8();
  float mrow[8], lrow[8];
#pragma unroll
  for (int r = 0; r < 8; ++r) { mrow[r] = -1e30f; lrow[r] = 0.f; }

  for (int kc = 0; kc < NT32; ++kc) {
    // ---- issue ALL fragment loads for this chunk up front ----
    ABFrag kf[4], vf[4];
    {
      int nt = kc * 2;
      const uint4* a0 = (const uint4*)&kbuf[(((size_t)(bh * 2 + 0) * NT16 + nt) << 9) + lane * 16];
      kf[0].u[0] = a0[0]; kf[0].u[1] = a0[1];
      const uint4* a1 = (const uint4*)&kbuf[(((size_t)(bh * 2 + 1) * NT16 + nt) << 9) + lane * 16];
      kf[1].u[0] = a1[0]; kf[1].u[1] = a1[1];
      const uint4* a2 = (const uint4*)&kbuf[(((size_t)(bh * 2 + 0) * NT16 + nt + 1) << 9) + lane * 16];
      kf[2].u[0] = a2[0]; kf[2].u[1] = a2[1];
      const uint4* a3 = (const uint4*)&kbuf[(((size_t)(bh * 2 + 1) * NT16 + nt + 1) << 9) + lane * 16];
      kf[3].u[0] = a3[0]; kf[3].u[1] = a3[1];
    }
#pragma unroll
    for (int dt = 0; dt < 4; ++dt) {
      const uint4* vp = (const uint4*)&vbuf[(((size_t)(bh * NT32 + kc) * 4 + dt) << 9) + lane * 16];
      vf[dt].u[0] = vp[0]; vf[dt].u[1] = vp[1];
    }

    // ---- S = Q K^T for 32 keys (two 16-col tiles), accumulated over D=64 ----
    v8f s0, s1;
    {
      v8f z0 = zero8();
      z0 = wmma_bf16(qf0, kf[0], z0);
      s0 = wmma_bf16(qf1, kf[1], z0);
      v8f z1 = zero8();
      z1 = wmma_bf16(qf0, kf[2], z1);
      s1 = wmma_bf16(qf1, kf[3], z1);
    }

    // ---- online softmax; rows live in 16-lane halves of the C layout ----
    float cm[8], mn[8], al[8];
#pragma unroll
    for (int r = 0; r < 8; ++r) cm[r] = fmaxf(s0[r], s1[r]);
    // batched butterfly row-max across the 16 lanes holding each row
#pragma unroll
    for (int r = 0; r < 8; ++r) cm[r] = fmaxf(cm[r], __shfl_xor(cm[r], 1, 32));
#pragma unroll
    for (int r = 0; r < 8; ++r) cm[r] = fmaxf(cm[r], __shfl_xor(cm[r], 2, 32));
#pragma unroll
    for (int r = 0; r < 8; ++r) cm[r] = fmaxf(cm[r], __shfl_xor(cm[r], 4, 32));
#pragma unroll
    for (int r = 0; r < 8; ++r) cm[r] = fmaxf(cm[r], __shfl_xor(cm[r], 8, 32));
#pragma unroll
    for (int r = 0; r < 8; ++r) {
      mn[r] = fmaxf(mrow[r], cm[r]);
      al[r] = __expf(mrow[r] - mn[r]);
      mrow[r] = mn[r];
    }
#pragma unroll
    for (int r = 0; r < 8; ++r) {
      float p0 = __expf(s0[r] - mn[r]);
      float p1 = __expf(s1[r] - mn[r]);
      int m = r + hi8;
      pb[m * 32 + nl]      = f2bf(p0);
      pb[m * 32 + 16 + nl] = f2bf(p1);
    }
#pragma unroll
    for (int dt = 0; dt < 4; ++dt)
#pragma unroll
      for (int r = 0; r < 8; ++r) o[dt][r] *= al[r];

    // wave-private LDS relayout: C-layout P -> A-fragment (DS ops are
    // in-order per wave; s_wait + clobber pins the compiler ordering)
    asm volatile("s_wait_dscnt 0" ::: "memory");
    ABFrag pf;
    const uint4* pp = (const uint4*)&pb[nl * 32 + hi8];
    pf.u[0] = pp[0]; pf.u[1] = pp[2];

    // ---- row-sum on the matrix pipe: rs = P x ones ----
    v8f rs = wmma_bf16(pf, ones, zero8());
#pragma unroll
    for (int r = 0; r < 8; ++r) lrow[r] = lrow[r] * al[r] + rs[r];

    // ---- O += P * V ----
#pragma unroll
    for (int dt = 0; dt < 4; ++dt) o[dt] = wmma_bf16(pf, vf[dt], o[dt]);
    asm volatile("s_wait_dscnt 0" ::: "memory");
  }

  // ---- normalize and emit bf16 row-major [TOK x CC] for the proj GEMM ----
  int tokbase = b * NN + mt * 16;
#pragma unroll
  for (int dt = 0; dt < 4; ++dt) {
    int col = h * 64 + dt * 16 + nl;
#pragma unroll
    for (int r = 0; r < 8; ++r) {
      float v = o[dt][r] / lrow[r];
      int tok = tokbase + r + hi8;
      attnb[(size_t)tok * CC + col] = f2bf(v);
    }
  }
}

// ---------------------------------------------------------------------------
extern "C" void kernel_launch(void* const* d_in, const int* in_sizes, int n_in,
                              void* d_out, int out_size, void* d_ws, size_t ws_size,
                              hipStream_t stream) {
  (void)in_sizes; (void)n_in; (void)out_size; (void)ws_size;
  const float* x      = (const float*)d_in[0];
  const float* qkv_w  = (const float*)d_in[1];
  const float* q_bias = (const float*)d_in[2];
  const float* v_bias = (const float*)d_in[3];
  const float* proj_w = (const float*)d_in[4];
  const float* proj_b = (const float*)d_in[5];
  float* out = (float*)d_out;

  char* ws = (char*)d_ws;
  size_t off = 0;
  auto carve = [&](size_t bytes) {
    void* p = ws + off;
    off += (bytes + 255) & ~(size_t)255;
    return p;
  };
  const size_t frag_elems = (size_t)BB * HH * NT16 * 2 * 512;  // 4,816,896
  unsigned short* xb    = (unsigned short*)carve((size_t)TOK * CC * 2);
  unsigned short* wqkv  = (unsigned short*)carve((size_t)CC * 3 * CC * 2);
  unsigned short* wp    = (unsigned short*)carve((size_t)CC * CC * 2);
  unsigned short* qa    = (unsigned short*)carve(frag_elems * 2);
  unsigned short* kb    = (unsigned short*)carve(frag_elems * 2);
  unsigned short* vb    = (unsigned short*)carve(frag_elems * 2);
  unsigned short* attnb = (unsigned short*)carve((size_t)TOK * CC * 2);

  // 1) convert x to bf16
  cvt_x_kernel<<<(TOK * CC) / 1024, 256, 0, stream>>>(x, xb, TOK * CC);
  // 2) pre-swizzle weights into WMMA B-fragment layouts
  pack_w_kernel<<<(CC * 3 * CC + 255) / 256, 256, 0, stream>>>(qkv_w, wqkv, CC, 3 * CC);
  pack_w_kernel<<<(CC * CC + 255) / 256, 256, 0, stream>>>(proj_w, wp, CC, CC);
  // 3) QKV GEMM, epilogue scatters Q(A-frag, pre-scaled), K^T(B-frag), V(B-frag)
  gemm_kernel<0><<<dim3((3 * CC) / 128, TOK / 128), 256, 0, stream>>>(
      xb, wqkv, 3 * CC, CC, q_bias, v_bias, nullptr, qa, kb, vb);
  // 4) flash attention: B*H*98 q-strips, 8 waves/block
  attn_kernel<<<(BB * HH * NT16) / 8, 256, 0, stream>>>(qa, kb, vb, attnb);
  // 5) output projection + bias -> fp32 out
  gemm_kernel<1><<<dim3(CC / 128, TOK / 128), 256, 0, stream>>>(
      attnb, wp, CC, CC, proj_b, nullptr, out, nullptr, nullptr, nullptr);
}
